// DynamicsModel_80882824118360
// MI455X (gfx1250) — compile-verified, matrix operand order checked by reading, not run on hardware
//
#include <hip/hip_runtime.h>
#include <hip/hip_bf16.h>

#define B_TOT 8192
#define KSLOT 7
#define DDIM  128
#define HIDD  256
#define ADIM  32
#define NACT  5
#define MROWS (B_TOT * KSLOT)   // 57344

typedef __attribute__((ext_vector_type(16))) __bf16 v16bf;
typedef __attribute__((ext_vector_type(8)))  float  v8f;

union FragAB { v16bf v; unsigned int u[8]; };

__device__ __forceinline__ unsigned short f2bf(float f) {
  unsigned int u = __float_as_uint(f);
  u += 0x7FFFu + ((u >> 16) & 1u);          // round-to-nearest-even
  return (unsigned short)(u >> 16);
}

// k-offset inside a 32-wide K window for dword r (0..7) and lane-half hh (0/1)
// per ISA 7.12.2: 16-bit A 16x32 layout
__device__ __forceinline__ int kmap(int r, int hh) {
  return (r < 4) ? (hh * 8 + r * 2) : (16 + hh * 8 + (r - 4) * 2);
}

// Load a 16x32 bf16 fragment from a row-major [rows][stride] bf16 buffer.
// Used for A (rows = M) and for B when weights are stored transposed [N][K].
__device__ __forceinline__ v16bf frag_load(const unsigned short* buf, int stride,
                                           int row0, int k0) {
  int lane = threadIdx.x & 31;
  int hh   = lane >> 4;
  int row  = row0 + (lane & 15);
  FragAB f;
#pragma unroll
  for (int r = 0; r < 8; ++r) {
    int k = k0 + kmap(r, hh);
    f.u[r] = *(const unsigned int*)(buf + (size_t)row * stride + k); // 2 bf16
  }
  return f.v;
}

__device__ __forceinline__ v8f frag_bias(const float* bias, int n0) {
  float bv = bias[n0 + (threadIdx.x & 15)];
  v8f c;
#pragma unroll
  for (int i = 0; i < 8; ++i) c[i] = bv;
  return c;
}

__device__ __forceinline__ v8f wmma_bf(v16bf a, v16bf b, v8f c) {
  return __builtin_amdgcn_wmma_f32_16x16x32_bf16(false, a, false, b,
                                                 (short)0, c, false, false);
}

__device__ __forceinline__ void frag_store_relu_bf16(unsigned short* buf, int stride,
                                                     int m0, int n0, v8f c) {
  int lane = threadIdx.x & 31;
  int n  = n0 + (lane & 15);
  int mb = m0 + 8 * (lane >> 4);
#pragma unroll
  for (int vv = 0; vv < 8; ++vv)
    buf[(size_t)(mb + vv) * stride + n] = f2bf(fmaxf(c[vv], 0.f));
}

__device__ __forceinline__ void frag_store_f32(float* buf, int stride,
                                               int m0, int n0, v8f c) {
  int lane = threadIdx.x & 31;
  int n  = n0 + (lane & 15);
  int mb = m0 + 8 * (lane >> 4);
#pragma unroll
  for (int vv = 0; vv < 8; ++vv)
    buf[(size_t)(mb + vv) * stride + n] = c[vv];
}

// ---------------------------------------------------------------------------
// K0a: action-encoder table: ae_table[a][o] for a in 0..4 (one-hot collapses)
// ---------------------------------------------------------------------------
__global__ void k_ae(const float* __restrict__ w1, const float* __restrict__ b1,
                     const float* __restrict__ w2, const float* __restrict__ b2,
                     float* __restrict__ aeTab) {
  int t = threadIdx.x;
  if (t >= NACT * ADIM) return;
  int a = t / ADIM, o = t % ADIM;
  float acc = b2[o];
  for (int j = 0; j < ADIM; ++j) {
    float h = fmaxf(w1[a * ADIM + j] + b1[j], 0.f);
    acc += h * w2[j * ADIM + o];
  }
  aeTab[t] = acc;
}

// ---------------------------------------------------------------------------
// K0b: transpose+convert f32 weight [din][dout] -> bf16 [dout][din]
// ---------------------------------------------------------------------------
__global__ void k_wt(const float* __restrict__ w, unsigned short* __restrict__ wt,
                     int din, int dout) {
  int id = blockIdx.x * blockDim.x + threadIdx.x;
  if (id >= din * dout) return;
  int n = id / din, k = id % din;
  wt[id] = f2bf(w[(size_t)k * dout + n]);
}

// ---------------------------------------------------------------------------
// K1: ObjectTransition fused MLP 160 -> 256 -> 256 -> 128 over M = B*K rows
// 64 rows / WG, 4 waves, one 16-row M-tile each. Activations in LDS bf16.
// ---------------------------------------------------------------------------
__global__ void __launch_bounds__(128) k1_object_transition(
    const float* __restrict__ slots, const int* __restrict__ action,
    const float* __restrict__ aeTab,
    const unsigned short* __restrict__ wt1, const float* __restrict__ b1,
    const unsigned short* __restrict__ wt2, const float* __restrict__ b2,
    const unsigned short* __restrict__ wt3, const float* __restrict__ b3,
    float* __restrict__ individual) {
  extern __shared__ __align__(16) char smem[];
  unsigned short* X  = (unsigned short*)smem;   // [64][160]
  unsigned short* H1 = X + 64 * 160;            // [64][256]
  unsigned short* H2 = H1 + 64 * 256;           // [64][256]
  const int m0 = blockIdx.x * 64;

  for (int idx = threadIdx.x; idx < 64 * 160; idx += 128) {
    int r = idx / 160, c = idx % 160;
    int m = m0 + r;
    float v;
    if (c < DDIM) v = slots[(size_t)m * DDIM + c];
    else          v = aeTab[action[m / KSLOT] * ADIM + (c - DDIM)];
    X[idx] = f2bf(v);
  }
  __syncthreads();

  const int mt = (threadIdx.x >> 5) * 16;
  const int lane = threadIdx.x & 31;

  // fc1: 160 -> 256, relu
  for (int nc = 0; nc < 4; ++nc) {
    v8f c[4];
#pragma unroll
    for (int j = 0; j < 4; ++j) c[j] = frag_bias(b1, (nc * 4 + j) * 16);
    for (int kt = 0; kt < 5; ++kt) {
      v16bf a = frag_load(X, 160, mt, kt * 32);
#pragma unroll
      for (int j = 0; j < 4; ++j)
        c[j] = wmma_bf(a, frag_load(wt1, 160, (nc * 4 + j) * 16, kt * 32), c[j]);
    }
#pragma unroll
    for (int j = 0; j < 4; ++j) frag_store_relu_bf16(H1, 256, mt, (nc * 4 + j) * 16, c[j]);
  }
  // fc2: 256 -> 256, relu (wave-private rows; no barrier needed)
  for (int nc = 0; nc < 4; ++nc) {
    v8f c[4];
#pragma unroll
    for (int j = 0; j < 4; ++j) c[j] = frag_bias(b2, (nc * 4 + j) * 16);
    for (int kt = 0; kt < 8; ++kt) {
      v16bf a = frag_load(H1, 256, mt, kt * 32);
#pragma unroll
      for (int j = 0; j < 4; ++j)
        c[j] = wmma_bf(a, frag_load(wt2, 256, (nc * 4 + j) * 16, kt * 32), c[j]);
    }
#pragma unroll
    for (int j = 0; j < 4; ++j) frag_store_relu_bf16(H2, 256, mt, (nc * 4 + j) * 16, c[j]);
  }
  // fc3: 256 -> 128 (linear) -> global f32
  for (int nc = 0; nc < 2; ++nc) {
    v8f c[4];
#pragma unroll
    for (int j = 0; j < 4; ++j) c[j] = frag_bias(b3, (nc * 4 + j) * 16);
    for (int kt = 0; kt < 8; ++kt) {
      v16bf a = frag_load(H2, 256, mt, kt * 32);
#pragma unroll
      for (int j = 0; j < 4; ++j)
        c[j] = wmma_bf(a, frag_load(wt3, 256, (nc * 4 + j) * 16, kt * 32), c[j]);
    }
#pragma unroll
    for (int j = 0; j < 4; ++j) {
      int n  = (nc * 4 + j) * 16 + (lane & 15);
      int mb = m0 + mt + 8 * (lane >> 4);
#pragma unroll
      for (int vv = 0; vv < 8; ++vv)
        individual[(size_t)(mb + vv) * DDIM + n] = c[j][vv];
    }
  }
}

// ---------------------------------------------------------------------------
// K2: attention q/k GEMMs + masked softmax -> wAttn[2][B][7][7]
// one wave per batch element, 4 elems / WG
// ---------------------------------------------------------------------------
#define K2_PER 37120  // per-wave LDS: Sb 4096 + Q 16384 + K 16384 + L 256

__global__ void __launch_bounds__(128) k2_attention(
    const float* __restrict__ slots,
    const unsigned short* __restrict__ wtq0, const float* __restrict__ bq0,
    const unsigned short* __restrict__ wtk0, const float* __restrict__ bk0,
    const unsigned short* __restrict__ wtq1, const float* __restrict__ bq1,
    const unsigned short* __restrict__ wtk1, const float* __restrict__ bk1,
    float* __restrict__ wAttn) {
  extern __shared__ __align__(16) char smem[];
  const int wv = threadIdx.x >> 5, lane = threadIdx.x & 31;
  char* base = smem + (size_t)wv * K2_PER;
  unsigned short* Sb = (unsigned short*)base;        // [16][128] bf16, rows>=7 zero
  float* Q  = (float*)(base + 16 * 128 * 2);         // [16][256]
  float* Kt = Q + 16 * 256;                          // [16][256]
  float* L  = Kt + 16 * 256;                         // [49] (+pad)
  const int b = blockIdx.x * 4 + wv;

  for (int idx = lane; idx < 16 * 128; idx += 32) {
    int r = idx >> 7, c = idx & 127;
    float v = (r < KSLOT) ? slots[((size_t)b * KSLOT + r) * DDIM + c] : 0.f;
    Sb[idx] = f2bf(v);
  }
  __syncthreads();

  for (int l = 0; l < 2; ++l) {
    const unsigned short* wq = l ? wtq1 : wtq0;
    const unsigned short* wk = l ? wtk1 : wtk0;
    const float* bq = l ? bq1 : bq0;
    const float* bk = l ? bk1 : bk0;

    v16bf a[4];
#pragma unroll
    for (int kt = 0; kt < 4; ++kt) a[kt] = frag_load(Sb, 128, 0, kt * 32);

    for (int nt = 0; nt < 16; ++nt) {
      v8f c = frag_bias(bq, nt * 16);
#pragma unroll
      for (int kt = 0; kt < 4; ++kt)
        c = wmma_bf(a[kt], frag_load(wq, 128, nt * 16, kt * 32), c);
      frag_store_f32(Q, 256, 0, nt * 16, c);

      c = frag_bias(bk, nt * 16);
#pragma unroll
      for (int kt = 0; kt < 4; ++kt)
        c = wmma_bf(a[kt], frag_load(wk, 128, nt * 16, kt * 32), c);
      frag_store_f32(Kt, 256, 0, nt * 16, c);
    }
    __syncthreads();

    for (int p = lane; p < 49; p += 32) {
      int i = p / 7, j = p % 7;
      float acc;
      if (i == j) acc = -1e9f;
      else {
        acc = 0.f;
        for (int h = 0; h < HIDD; ++h) acc += Q[i * 256 + h] * Kt[j * 256 + h];
        acc *= 0.0625f;                     // HID^-0.5 = 1/16
      }
      L[p] = acc;
    }
    __syncthreads();

    if (lane < KSLOT) {
      float mx = -1e30f;
      for (int j = 0; j < 7; ++j) mx = fmaxf(mx, L[lane * 7 + j]);
      float e[7], s = 0.f;
      for (int j = 0; j < 7; ++j) { e[j] = expf(L[lane * 7 + j] - mx); s += e[j]; }
      float inv = 1.f / s;
      for (int j = 0; j < 7; ++j)
        wAttn[((size_t)l * B_TOT + b) * 49 + lane * 7 + j] = e[j] * inv;
    }
    __syncthreads();
  }
}

// ---------------------------------------------------------------------------
// K3: pairwise MLP (2D -> HID -> D) + attention-weighted sum.
// Weights parked in LDS (~192KB bf16) and reused across 8 batch elems / WG.
// Pair rows built on the fly from the 7x128 slot tile (row = i*7+j).
// ---------------------------------------------------------------------------
__global__ void __launch_bounds__(128) k3_pairwise(
    const float* __restrict__ slots,
    const unsigned short* __restrict__ wth1_0, const float* __restrict__ bh1_0,
    const unsigned short* __restrict__ wth2_0, const float* __restrict__ bh2_0,
    const unsigned short* __restrict__ wth1_1, const float* __restrict__ bh1_1,
    const unsigned short* __restrict__ wth2_1, const float* __restrict__ bh2_1,
    const float* __restrict__ wAttn, float* __restrict__ interTot) {
  extern __shared__ __align__(16) char smem[];
  unsigned short* WH1 = (unsigned short*)smem;       // [256][256] bf16  131072 B
  unsigned short* WH2 = WH1 + 256 * 256;             // [128][256] bf16   65536 B
  unsigned short* H1  = WH2 + 128 * 256;             // [64][256]  bf16   32768 B
  unsigned short* S   = H1 + 64 * 256;               // [7][128]   bf16 (+pad)
  float* H2 = (float*)((char*)S + 2048);             // [64][128]  f32    32768 B
  float* WA = H2 + 64 * 128;                         // [49]

  const int b0 = blockIdx.x * 8;
  const int wv = threadIdx.x >> 5, lane = threadIdx.x & 31;
  const int m0 = wv * 16;
  const int row = m0 + (lane & 15);
  const bool valid = row < 49;
  const int ii = valid ? row / 7 : 0;
  const int jj = valid ? row - ii * 7 : 0;
  const int hh = lane >> 4;

  for (int l = 0; l < 2; ++l) {
    const unsigned short* wh1 = l ? wth1_1 : wth1_0;
    const unsigned short* wh2 = l ? wth2_1 : wth2_0;
    const float* bb1 = l ? bh1_1 : bh1_0;
    const float* bb2 = l ? bh2_1 : bh2_0;

    __syncthreads();
    {   // park this layer's weights in LDS (dword copies)
      const unsigned int* s1 = (const unsigned int*)wh1;
      unsigned int* d1 = (unsigned int*)WH1;
      for (int i = threadIdx.x; i < 256 * 256 / 2; i += 128) d1[i] = s1[i];
      const unsigned int* s2 = (const unsigned int*)wh2;
      unsigned int* d2 = (unsigned int*)WH2;
      for (int i = threadIdx.x; i < 128 * 256 / 2; i += 128) d2[i] = s2[i];
    }
    __syncthreads();

    for (int bb = 0; bb < 8; ++bb) {
      const int b = b0 + bb;
      for (int idx = threadIdx.x; idx < KSLOT * DDIM; idx += 128)
        S[idx] = f2bf(slots[(size_t)b * KSLOT * DDIM + idx]);
      for (int idx = threadIdx.x; idx < 49; idx += 128)
        WA[idx] = wAttn[((size_t)l * B_TOT + b) * 49 + idx];
      __syncthreads();

      // h_fc1: pair(256) -> 256, relu.  A built from S: k<128 -> slots[i], else slots[j]
      for (int nc = 0; nc < 4; ++nc) {
        v8f c[4];
#pragma unroll
        for (int j = 0; j < 4; ++j) c[j] = frag_bias(bb1, (nc * 4 + j) * 16);
        for (int kt = 0; kt < 8; ++kt) {
          FragAB f;
#pragma unroll
          for (int r = 0; r < 8; ++r) {
            int k = kt * 32 + kmap(r, hh);
            unsigned int u = 0;
            if (valid) {
              const unsigned short* src = (k < DDIM) ? (S + ii * DDIM + k)
                                                     : (S + jj * DDIM + (k - DDIM));
              u = *(const unsigned int*)src;
            }
            f.u[r] = u;
          }
#pragma unroll
          for (int j = 0; j < 4; ++j)
            c[j] = wmma_bf(f.v, frag_load(WH1, 256, (nc * 4 + j) * 16, kt * 32), c[j]);
        }
#pragma unroll
        for (int j = 0; j < 4; ++j) frag_store_relu_bf16(H1, 256, m0, (nc * 4 + j) * 16, c[j]);
      }
      // h_fc2: 256 -> 128 (linear) -> H2 f32
      for (int nc = 0; nc < 2; ++nc) {
        v8f c[4];
#pragma unroll
        for (int j = 0; j < 4; ++j) c[j] = frag_bias(bb2, (nc * 4 + j) * 16);
        for (int kt = 0; kt < 8; ++kt) {
          v16bf a = frag_load(H1, 256, m0, kt * 32);
#pragma unroll
          for (int j = 0; j < 4; ++j)
            c[j] = wmma_bf(a, frag_load(WH2, 256, (nc * 4 + j) * 16, kt * 32), c[j]);
        }
#pragma unroll
        for (int j = 0; j < 4; ++j) frag_store_f32(H2, 128, m0, (nc * 4 + j) * 16, c[j]);
      }
      __syncthreads();

      // inter[b,i,:] += sum_j w[i,j] * h[i*7+j,:]   (diag weight ~0 from softmax)
      for (int idx = threadIdx.x; idx < KSLOT * DDIM; idx += 128) {
        int i = idx >> 7, d = idx & 127;
        float acc = 0.f;
#pragma unroll
        for (int j = 0; j < 7; ++j) acc += WA[i * 7 + j] * H2[(i * 7 + j) * DDIM + d];
        size_t o = ((size_t)b * KSLOT + i) * DDIM + d;
        if (l == 0) interTot[o] = acc;
        else        interTot[o] += acc;   // same WG handles b in both layers: safe
      }
      __syncthreads();
    }
  }
}

// ---------------------------------------------------------------------------
// K4: combine + LayerNorm + out MLP 128 -> 256 -> 128 + residual + constraints
// ---------------------------------------------------------------------------
__global__ void __launch_bounds__(128) k4_output(
    const float* __restrict__ slots,
    const float* __restrict__ individual, const float* __restrict__ interTot,
    const float* __restrict__ lnS, const float* __restrict__ lnB,
    const unsigned short* __restrict__ wto1, const float* __restrict__ bo1,
    const unsigned short* __restrict__ wto2, const float* __restrict__ bo2,
    float* __restrict__ out) {
  extern __shared__ __align__(16) char smem[];
  float* C64 = (float*)smem;                           // [64][128] f32
  unsigned short* XN = (unsigned short*)(C64 + 64 * 128); // [64][128] bf16
  unsigned short* H  = XN + 64 * 128;                  // [64][256] bf16
  const int m0 = blockIdx.x * 64;

  for (int idx = threadIdx.x; idx < 64 * DDIM; idx += 128) {
    size_t g = (size_t)m0 * DDIM + idx;
    C64[idx] = individual[g] + interTot[g];
  }
  __syncthreads();

  if (threadIdx.x < 64) {
    int r = threadIdx.x;
    float mu = 0.f;
    for (int d = 0; d < DDIM; ++d) mu += C64[r * DDIM + d];
    mu *= (1.f / DDIM);
    float var = 0.f;
    for (int d = 0; d < DDIM; ++d) { float t = C64[r * DDIM + d] - mu; var += t * t; }
    var *= (1.f / DDIM);
    float rs = rsqrtf(var + 1e-6f);
    for (int d = 0; d < DDIM; ++d)
      XN[r * DDIM + d] = f2bf((C64[r * DDIM + d] - mu) * rs * lnS[d] + lnB[d]);
  }
  __syncthreads();

  const int mt = (threadIdx.x >> 5) * 16;
  const int lane = threadIdx.x & 31;

  // out_fc1: 128 -> 256, relu
  for (int nc = 0; nc < 4; ++nc) {
    v8f c[4];
#pragma unroll
    for (int j = 0; j < 4; ++j) c[j] = frag_bias(bo1, (nc * 4 + j) * 16);
    for (int kt = 0; kt < 4; ++kt) {
      v16bf a = frag_load(XN, 128, mt, kt * 32);
#pragma unroll
      for (int j = 0; j < 4; ++j)
        c[j] = wmma_bf(a, frag_load(wto1, 128, (nc * 4 + j) * 16, kt * 32), c[j]);
    }
#pragma unroll
    for (int j = 0; j < 4; ++j) frag_store_relu_bf16(H, 256, mt, (nc * 4 + j) * 16, c[j]);
  }
  // out_fc2: 256 -> 128 + residual + constraints
  for (int nc = 0; nc < 2; ++nc) {
    v8f c[4];
#pragma unroll
    for (int j = 0; j < 4; ++j) c[j] = frag_bias(bo2, (nc * 4 + j) * 16);
    for (int kt = 0; kt < 8; ++kt) {
      v16bf a = frag_load(H, 256, mt, kt * 32);
#pragma unroll
      for (int j = 0; j < 4; ++j)
        c[j] = wmma_bf(a, frag_load(wto2, 256, (nc * 4 + j) * 16, kt * 32), c[j]);
    }
#pragma unroll
    for (int j = 0; j < 4; ++j) {
      int n  = (nc * 4 + j) * 16 + (lane & 15);
      int mb = m0 + mt + 8 * (lane >> 4);
#pragma unroll
      for (int vv = 0; vv < 8; ++vv) {
        size_t g = (size_t)(mb + vv) * DDIM + n;
        float val = slots[g] + c[j][vv];
        if      (n < 2) val = fminf(fmaxf(val, -2.f), 2.f);
        else if (n < 4) val = fminf(fmaxf(val, -1.f), 1.f);
        else if (n < 6) val = 0.5f * (fmaxf(val, 0.f) + log1pf(expf(-fabsf(val))));
        out[g] = val;
      }
    }
  }
}

// ---------------------------------------------------------------------------
extern "C" void kernel_launch(void* const* d_in, const int* in_sizes, int n_in,
                              void* d_out, int out_size, void* d_ws, size_t ws_size,
                              hipStream_t stream) {
  (void)in_sizes; (void)n_in; (void)out_size; (void)ws_size;
  const float* slots  = (const float*)d_in[0];
  const int*   action = (const int*)d_in[1];
  const float* ae1w = (const float*)d_in[2];  const float* ae1b = (const float*)d_in[3];
  const float* ae2w = (const float*)d_in[4];  const float* ae2b = (const float*)d_in[5];
  const float* ot1w = (const float*)d_in[6];  const float* ot1b = (const float*)d_in[7];
  const float* ot2w = (const float*)d_in[8];  const float* ot2b = (const float*)d_in[9];
  const float* ot3w = (const float*)d_in[10]; const float* ot3b = (const float*)d_in[11];
  const float* h1w0 = (const float*)d_in[12]; const float* h1b0 = (const float*)d_in[13];
  const float* h2w0 = (const float*)d_in[14]; const float* h2b0 = (const float*)d_in[15];
  const float* qw0  = (const float*)d_in[16]; const float* qb0  = (const float*)d_in[17];
  const float* kw0  = (const float*)d_in[18]; const float* kb0  = (const float*)d_in[19];
  const float* h1w1 = (const float*)d_in[20]; const float* h1b1 = (const float*)d_in[21];
  const float* h2w1 = (const float*)d_in[22]; const float* h2b1 = (const float*)d_in[23];
  const float* qw1  = (const float*)d_in[24]; const float* qb1  = (const float*)d_in[25];
  const float* kw1  = (const float*)d_in[26]; const float* kb1  = (const float*)d_in[27];
  const float* lnS  = (const float*)d_in[28]; const float* lnB  = (const float*)d_in[29];
  const float* o1w  = (const float*)d_in[30]; const float* o1b  = (const float*)d_in[31];
  const float* o2w  = (const float*)d_in[32]; const float* o2b  = (const float*)d_in[33];
  float* out = (float*)d_out;

  char* ws = (char*)d_ws;
  size_t off = 0;
  auto alloc = [&](size_t bytes) { size_t o = off; off = (off + bytes + 255) & ~(size_t)255; return o; };

  float*          aeTab   = (float*)(ws + alloc(NACT * ADIM * 4));
  unsigned short* wt_ot1  = (unsigned short*)(ws + alloc(256 * 160 * 2));
  unsigned short* wt_ot2  = (unsigned short*)(ws + alloc(256 * 256 * 2));
  unsigned short* wt_ot3  = (unsigned short*)(ws + alloc(128 * 256 * 2));
  unsigned short* wt_h1_0 = (unsigned short*)(ws + alloc(256 * 256 * 2));
  unsigned short* wt_h2_0 = (unsigned short*)(ws + alloc(128 * 256 * 2));
  unsigned short* wt_q_0  = (unsigned short*)(ws + alloc(256 * 128 * 2));
  unsigned short* wt_k_0  = (unsigned short*)(ws + alloc(256 * 128 * 2));
  unsigned short* wt_h1_1 = (unsigned short*)(ws + alloc(256 * 256 * 2));
  unsigned short* wt_h2_1 = (unsigned short*)(ws + alloc(128 * 256 * 2));
  unsigned short* wt_q_1  = (unsigned short*)(ws + alloc(256 * 128 * 2));
  unsigned short* wt_k_1  = (unsigned short*)(ws + alloc(256 * 128 * 2));
  unsigned short* wt_o1   = (unsigned short*)(ws + alloc(256 * 128 * 2));
  unsigned short* wt_o2   = (unsigned short*)(ws + alloc(128 * 256 * 2));
  float* individual = (float*)(ws + alloc((size_t)MROWS * DDIM * 4));
  float* interTot   = (float*)(ws + alloc((size_t)MROWS * DDIM * 4));
  float* wAttn      = (float*)(ws + alloc((size_t)2 * B_TOT * 49 * 4));

  hipLaunchKernelGGL(k_ae, dim3(1), dim3(192), 0, stream, ae1w, ae1b, ae2w, ae2b, aeTab);

  auto wt = [&](const float* w, unsigned short* t, int din, int dout) {
    int tot = din * dout;
    hipLaunchKernelGGL(k_wt, dim3((tot + 255) / 256), dim3(256), 0, stream, w, t, din, dout);
  };
  wt(ot1w, wt_ot1, 160, 256);  wt(ot2w, wt_ot2, 256, 256);  wt(ot3w, wt_ot3, 256, 128);
  wt(h1w0, wt_h1_0, 256, 256); wt(h2w0, wt_h2_0, 256, 128);
  wt(qw0,  wt_q_0, 128, 256);  wt(kw0,  wt_k_0, 128, 256);
  wt(h1w1, wt_h1_1, 256, 256); wt(h2w1, wt_h2_1, 256, 128);
  wt(qw1,  wt_q_1, 128, 256);  wt(kw1,  wt_k_1, 128, 256);
  wt(o1w,  wt_o1, 128, 256);   wt(o2w,  wt_o2, 256, 128);

  // K1: 57344 rows, 64 per WG
  size_t sh1 = (size_t)64 * 160 * 2 + 64 * 256 * 2 + 64 * 256 * 2;   // 86016
  hipLaunchKernelGGL(k1_object_transition, dim3(MROWS / 64), dim3(128), sh1, stream,
                     slots, action, aeTab, wt_ot1, ot1b, wt_ot2, ot2b, wt_ot3, ot3b,
                     individual);

  // K2: one wave per b, 4 b per WG
  size_t sh2 = (size_t)K2_PER * 4;                                   // 148480
  hipLaunchKernelGGL(k2_attention, dim3(B_TOT / 4), dim3(128), sh2, stream,
                     slots, wt_q_0, qb0, wt_k_0, kb0, wt_q_1, qb1, wt_k_1, kb1, wAttn);

  // K3: 8 b per WG, weights resident in LDS
  size_t sh3 = 131072 + 65536 + 32768 + 2048 + 32768 + 256;          // 264448
  hipLaunchKernelGGL(k3_pairwise, dim3(B_TOT / 8), dim3(128), sh3, stream,
                     slots, wt_h1_0, h1b0, wt_h2_0, h2b0, wt_h1_1, h1b1, wt_h2_1, h2b1,
                     wAttn, interTot);

  // K4: 64 rows per WG
  size_t sh4 = (size_t)64 * 128 * 4 + 64 * 128 * 2 + 64 * 256 * 2;   // 81920
  hipLaunchKernelGGL(k4_output, dim3(MROWS / 64), dim3(128), sh4, stream,
                     slots, individual, interTot, lnS, lnB, wt_o1, o1b, wt_o2, o2b, out);
}
